// TransformerBlockWithSSM_11708080849428
// MI455X (gfx1250) — matrix operand order repaired and checked by reading, not compiled, not run on hardware
//
#include <hip/hip_runtime.h>
#include <math.h>

// ---------------------------------------------------------------------------
// Problem dimensions
// ---------------------------------------------------------------------------
#define BATCH   4
#define LSEQ    2048
#define DDIM    1024
#define EDIM    2048
#define NSTATE  16
#define KCONV   4
#define DTRANK  64
#define HDIM    4096
#define PROJW   96
#define TOK     (BATCH*LSEQ)

// ---------------------------------------------------------------------------
// Vector types
// ---------------------------------------------------------------------------
typedef _Float16 v16h  __attribute__((ext_vector_type(16)));
typedef _Float16 half8 __attribute__((ext_vector_type(8)));
typedef float    v8f   __attribute__((ext_vector_type(8)));
typedef unsigned int u32x4 __attribute__((ext_vector_type(4)));
typedef int      i32x8 __attribute__((ext_vector_type(8)));
typedef int      i32x4 __attribute__((ext_vector_type(4)));

union AFrag { v16h v; half8 h[2]; };

// ---------------------------------------------------------------------------
// TDM availability / arity detection (ROCm 7.2 = 5 args; therock = 6 args)
// ---------------------------------------------------------------------------
#if defined(__has_builtin)
#  if __has_builtin(__builtin_amdgcn_tensor_load_to_lds)
#    define HAVE_TDM 1
#  endif
#endif
#ifndef HAVE_TDM
#  define HAVE_TDM 0
#endif
#if __has_include(<hip/amd_detail/amd_gfx1250_TDM.h>)
#  define TDM_SIX_ARGS 1
#else
#  define TDM_SIX_ARGS 0
#endif

#if HAVE_TDM
// 2D f16 tile load via Tensor Data Mover (ISA 08_async_tensor §8).
// Writes tile (tile1 rows x tile0 halfs) to LDS with 16B padding after every
// 64B row chunk -> LDS row stride = 80B = 40 halfs (matches fragment layout).
// OOB rows/cols (beyond td0/td1) are zero-filled by the TDM.
__device__ __forceinline__ void tdm_load_2d_f16(const void* gaddr, unsigned lds_off,
                                                unsigned td0, unsigned td1,
                                                unsigned long long stride0,
                                                unsigned tile0, unsigned tile1)
{
    unsigned long long ga = (unsigned long long)(size_t)gaddr;
    u32x4 g0;
    g0[0] = 1u;                                            // count=1, user mode
    g0[1] = lds_off;                                       // LDS byte address
    g0[2] = (unsigned)ga;                                  // global_addr[31:0]
    g0[3] = (unsigned)((ga >> 32) & 0x01FFFFFFull) | (2u << 30); // addr[56:32] | type=2
    i32x8 g1;
    g1[0] = (int)((1u << 16)      // data_size = 1 (2 bytes)
                | (1u << 20)      // pad_enable
                | (3u << 22)      // pad_interval: 16 DWORDs (64B)
                | (3u << 25));    // pad_amount:   4 DWORDs (16B)
    g1[1] = (int)((td0 & 0xFFFFu) << 16);                  // tensor_dim0[15:0] @bit48
    g1[2] = (int)((td0 >> 16) | ((td1 & 0xFFFFu) << 16));  // dim0 hi | dim1 lo
    g1[3] = (int)((td1 >> 16) | (tile0 << 16));            // dim1 hi | tile_dim0
    g1[4] = (int)(tile1 & 0xFFFFu);                        // tile_dim1 (tile_dim2=0)
    g1[5] = (int)(unsigned)(stride0 & 0xFFFFFFFFull);      // tensor_dim0_stride lo
    g1[6] = (int)(unsigned)((stride0 >> 32) & 0xFFFFull);  // stride hi (stride1=0)
    g1[7] = 0;
    i32x4 z4; z4[0] = 0; z4[1] = 0; z4[2] = 0; z4[3] = 0;
#if TDM_SIX_ARGS
    i32x8 z8;
    for (int j = 0; j < 8; ++j) z8[j] = 0;
    __builtin_amdgcn_tensor_load_to_lds(g0, g1, z4, z4, z8, 0);
#else
    __builtin_amdgcn_tensor_load_to_lds(g0, g1, z4, z4, 0);
#endif
}
#endif

// ---------------------------------------------------------------------------
// Fused epilogues
//  0: none   1: +bias   2: +bias,gelu(erf)   3: +bias,softplus   4: +residual
// ---------------------------------------------------------------------------
template<int EPI>
__device__ __forceinline__ float apply_epi(float a, float b, float r) {
    if (EPI == 0) return a;
    if (EPI == 1) return a + b;
    if (EPI == 2) { float t = a + b; return 0.5f * t * (1.0f + erff(t * 0.7071067811865476f)); }
    if (EPI == 3) { float t = a + b; return (t > 20.0f) ? t : log1pf(__expf(t)); }
    /* EPI == 4 */ return a + r;
}

// ---------------------------------------------------------------------------
// WMMA GEMM:  C[M,N] = epi(A[M,K] @ Bt[N,K]^T)
//   A:  f16 row-major (lda), Bt: f16 pre-transposed (row = output col, ld = K)
//   128x128x32 block tile, 8 waves, wave tile 32x64 (2x4 WMMA accumulators).
//   Tiles staged by the Tensor Data Mover (double-buffered LDS) when available.
// ---------------------------------------------------------------------------
#define BM 128
#define BN 128
#define BK 32
#define ASL 40   // padded LDS row stride in halfs (64B data + 16B pad)

template<int EPI, typename OutT>
__global__ __launch_bounds__(256)
void gemm_wmma_kernel(const _Float16* __restrict__ A, const _Float16* __restrict__ Bt,
                      const float* __restrict__ bias, const float* __restrict__ resid,
                      OutT* __restrict__ C,
                      int M, int N, int K, int lda, int ldc)
{
    __shared__ _Float16 As[2][BM * ASL];
    __shared__ _Float16 Bs[2][BN * ASL];

    const int tid  = threadIdx.x;
    const int lane = tid & 31;
    const int wid  = tid >> 5;
    const int wm   = wid & 3;          // 4 waves x 32 rows
    const int wn   = wid >> 2;         // 2 waves x 64 cols
    const int lh   = lane >> 4;
    const int ll   = lane & 15;
    const int blockM = blockIdx.y * BM;
    const int blockN = blockIdx.x * BN;

    v8f acc[2][4];
    #pragma unroll
    for (int tm = 0; tm < 2; ++tm)
        #pragma unroll
        for (int tn = 0; tn < 4; ++tn)
            #pragma unroll
            for (int v = 0; v < 8; ++v) acc[tm][tn][v] = 0.0f;

#if HAVE_TDM
    if (wid == 0) {   // one TDM pair per workgroup, wave 0 issues
        tdm_load_2d_f16(A + (size_t)blockM * lda, (unsigned)(size_t)&As[0][0],
                        (unsigned)K, (unsigned)(M - blockM),
                        (unsigned long long)lda, BK, BM);
        tdm_load_2d_f16(Bt + (size_t)blockN * K, (unsigned)(size_t)&Bs[0][0],
                        (unsigned)K, (unsigned)(N - blockN),
                        (unsigned long long)K, BK, BN);
    }
#endif

    for (int kk = 0; kk < K; kk += BK) {
        const int buf = (kk >> 5) & 1;
#if HAVE_TDM
        if (wid == 0) __builtin_amdgcn_s_wait_tensorcnt(0);
        __syncthreads();
        if (wid == 0 && kk + BK < K) {   // prefetch next tiles into other buffer
            const int kn = kk + BK;
            tdm_load_2d_f16(A + (size_t)blockM * lda + kn, (unsigned)(size_t)&As[buf ^ 1][0],
                            (unsigned)(K - kn), (unsigned)(M - blockM),
                            (unsigned long long)lda, BK, BM);
            tdm_load_2d_f16(Bt + (size_t)blockN * K + kn, (unsigned)(size_t)&Bs[buf ^ 1][0],
                            (unsigned)(K - kn), (unsigned)(N - blockN),
                            (unsigned long long)K, BK, BN);
        }
#else
        __syncthreads();
        // cooperative staging fallback: 512 x half8 chunks per tile
        #pragma unroll
        for (int p = 0; p < 2; ++p) {
            int chunk = tid + p * 256;
            int r  = chunk >> 2;
            int c8 = (chunk & 3) * 8;
            half8 av = *(const half8*)(A + (size_t)(blockM + r) * lda + kk + c8);
            *(half8*)&As[buf][r * ASL + c8] = av;
            half8 bv;
            if (blockN + r < N) {
                bv = *(const half8*)(Bt + (size_t)(blockN + r) * K + kk + c8);
            } else {
                #pragma unroll
                for (int j = 0; j < 8; ++j) bv[j] = (_Float16)0.0f;
            }
            *(half8*)&Bs[buf][r * ASL + c8] = bv;
        }
        __syncthreads();
#endif
        // fragment loads (ISA 7.12.2 layouts), two 16B DS loads each
        AFrag af[2], bf[4];
        #pragma unroll
        for (int tm = 0; tm < 2; ++tm) {
            int m = wm * 32 + tm * 16 + ll;
            const _Float16* p = &As[buf][m * ASL + lh * 8];   // k = lh*8+i, then +16
            af[tm].h[0] = *(const half8*)(p);
            af[tm].h[1] = *(const half8*)(p + 16);
        }
        #pragma unroll
        for (int tn = 0; tn < 4; ++tn) {
            int n = wn * 64 + tn * 16 + ll;
            const _Float16* p = &Bs[buf][n * ASL + lh * 8];   // k = lh*8+i, then +16
            bf[tn].h[0] = *(const half8*)(p);
            bf[tn].h[1] = *(const half8*)(p + 16);
        }
        #pragma unroll
        for (int tm = 0; tm < 2; ++tm)
            #pragma unroll
            for (int tn = 0; tn < 4; ++tn)
                acc[tm][tn] = __builtin_amdgcn_wmma_f32_16x16x32_f16(
                    false, af[tm].v, false, bf[tn].v,
                    (short)0, acc[tm][tn], false, false);
    }

    // epilogue + store (C layout: vgpr v -> m = v + 8*lh, n = ll)
    #pragma unroll
    for (int tm = 0; tm < 2; ++tm) {
        int m0 = blockM + wm * 32 + tm * 16 + lh * 8;
        #pragma unroll
        for (int tn = 0; tn < 4; ++tn) {
            int n = blockN + wn * 64 + tn * 16 + ll;
            if (n < N) {
                float bv = (EPI == 1 || EPI == 2 || EPI == 3) ? bias[n] : 0.0f;
                #pragma unroll
                for (int v = 0; v < 8; ++v) {
                    int m = m0 + v;
                    float rv = (EPI == 4) ? resid[(size_t)m * ldc + n] : 0.0f;
                    C[(size_t)m * ldc + n] = (OutT)apply_epi<EPI>(acc[tm][tn][v], bv, rv);
                }
            }
        }
    }
}

// ---------------------------------------------------------------------------
// f32 -> f16 convert / transpose-convert (weight pre-pass, x pre-pass)
// ---------------------------------------------------------------------------
__global__ __launch_bounds__(256)
void cvt_f16_kernel(const float* __restrict__ in, _Float16* __restrict__ out, int n)
{
    int i = blockIdx.x * 256 + threadIdx.x;
    if (i < n) out[i] = (_Float16)in[i];
}

__global__ __launch_bounds__(256)
void cvt_transpose_f16_kernel(const float* __restrict__ in, _Float16* __restrict__ out,
                              int R, int Cc)   // in: R x Cc, out: Cc x R
{
    int i = blockIdx.x * 256 + threadIdx.x;
    if (i < R * Cc) {
        int r = i / Cc, c = i % Cc;
        out[(size_t)c * R + r] = (_Float16)in[i];
    }
}

// ---------------------------------------------------------------------------
// Depthwise causal conv1d (K=4) + SiLU. xp = xz[:, 0:E] (f16 in/out, f32 math)
// ---------------------------------------------------------------------------
__global__ __launch_bounds__(256)
void conv_silu_kernel(const _Float16* __restrict__ xz, const float* __restrict__ w,
                      const float* __restrict__ cb, _Float16* __restrict__ xc)
{
    size_t idx = (size_t)blockIdx.x * 256 + threadIdx.x;
    if (idx >= (size_t)BATCH * LSEQ * EDIM) return;
    int e     = (int)(idx % EDIM);
    size_t bl = idx / EDIM;
    int l     = (int)(bl % LSEQ);
    size_t b  = bl / LSEQ;

    float acc = cb[e];
    #pragma unroll
    for (int j = 0; j < KCONV; ++j) {
        int ls = l - (KCONV - 1) + j;
        if (ls >= 0)
            acc += w[e * KCONV + j] * (float)xz[(b * LSEQ + (size_t)ls) * (2 * EDIM) + e];
    }
    xc[idx] = (_Float16)(acc / (1.0f + __expf(-acc)));
}

// ---------------------------------------------------------------------------
// Selective scan: thread = (b,e), 16-state recurrence in registers,
// B_t/C_t broadcast via LDS. Fuses +D*u and SiLU(z) gate.
// ---------------------------------------------------------------------------
__global__ __launch_bounds__(256)
void ssm_scan_kernel(const _Float16* __restrict__ dt, const _Float16* __restrict__ proj,
                     const _Float16* __restrict__ xc, const _Float16* __restrict__ xz,
                     const float* __restrict__ A_log, const float* __restrict__ Dp,
                     _Float16* __restrict__ Y)
{
    const int blocksPerB = EDIM / 256;
    const int b = blockIdx.x / blocksPerB;
    const int e = (blockIdx.x % blocksPerB) * 256 + threadIdx.x;

    __shared__ float sB[NSTATE], sC[NSTATE];

    float negA[NSTATE], h[NSTATE];
    #pragma unroll
    for (int n = 0; n < NSTATE; ++n) {
        negA[n] = -__expf(A_log[(size_t)e * NSTATE + n]);
        h[n] = 0.0f;
    }
    const float dpar = Dp[e];

    for (int l = 0; l < LSEQ; ++l) {
        size_t row = (size_t)b * LSEQ + l;
        if (threadIdx.x < NSTATE)
            sB[threadIdx.x] = (float)proj[row * PROJW + DTRANK + threadIdx.x];
        else if (threadIdx.x < 2 * NSTATE)
            sC[threadIdx.x - NSTATE] = (float)proj[row * PROJW + DTRANK + NSTATE + (threadIdx.x - NSTATE)];
        __syncthreads();

        float dtv = (float)dt[row * EDIM + e];
        float u   = (float)xc[row * EDIM + e];
        float du  = dtv * u;
        float y   = 0.0f;
        #pragma unroll
        for (int n = 0; n < NSTATE; ++n) {
            float dA = __expf(dtv * negA[n]);
            h[n] = dA * h[n] + du * sB[n];
            y += h[n] * sC[n];
        }
        y += dpar * u;
        float zv  = (float)xz[row * (2 * EDIM) + EDIM + e];
        float sil = zv / (1.0f + __expf(-zv));
        Y[row * EDIM + e] = (_Float16)(y * sil);
        __syncthreads();
    }
}

// ---------------------------------------------------------------------------
// LayerNorm over D=1024 per token (f32 in, f16 out for next GEMM)
// ---------------------------------------------------------------------------
__global__ __launch_bounds__(256)
void layernorm_kernel(const float* __restrict__ Hb, const float* __restrict__ w,
                      const float* __restrict__ bias, _Float16* __restrict__ out)
{
    const int row = blockIdx.x;
    const int tid = threadIdx.x;
    const float* p = Hb + (size_t)row * DDIM;

    float s = 0.0f, s2 = 0.0f;
    #pragma unroll
    for (int i = tid; i < DDIM; i += 256) { float v = p[i]; s += v; s2 += v * v; }

    __shared__ float rs[256], rs2[256];
    rs[tid] = s; rs2[tid] = s2;
    __syncthreads();
    for (int off = 128; off > 0; off >>= 1) {
        if (tid < off) { rs[tid] += rs[tid + off]; rs2[tid] += rs2[tid + off]; }
        __syncthreads();
    }
    float mu   = rs[0] * (1.0f / DDIM);
    float var  = rs2[0] * (1.0f / DDIM) - mu * mu;
    float rstd = rsqrtf(var + 1e-5f);
    #pragma unroll
    for (int i = tid; i < DDIM; i += 256)
        out[(size_t)row * DDIM + i] = (_Float16)((p[i] - mu) * rstd * w[i] + bias[i]);
}

// ---------------------------------------------------------------------------
// Launcher
// ---------------------------------------------------------------------------
extern "C" void kernel_launch(void* const* d_in, const int* in_sizes, int n_in,
                              void* d_out, int out_size, void* d_ws, size_t ws_size,
                              hipStream_t stream)
{
    (void)in_sizes; (void)n_in; (void)out_size; (void)ws_size;

    const float* x       = (const float*)d_in[0];
    const float* W_in    = (const float*)d_in[1];
    const float* conv_w  = (const float*)d_in[2];
    const float* conv_b  = (const float*)d_in[3];
    const float* W_xproj = (const float*)d_in[4];
    const float* W_dt    = (const float*)d_in[5];
    const float* b_dt    = (const float*)d_in[6];
    const float* A_log   = (const float*)d_in[7];
    const float* D_param = (const float*)d_in[8];
    const float* W_out   = (const float*)d_in[9];
    const float* ln_w    = (const float*)d_in[10];
    const float* ln_b    = (const float*)d_in[11];
    const float* W1      = (const float*)d_in[12];
    const float* b1      = (const float*)d_in[13];
    const float* W2      = (const float*)d_in[14];
    const float* b2      = (const float*)d_in[15];
    float* out = (float*)d_out;

    // Workspace layout (bytes)
    char* ws = (char*)d_ws;
    _Float16* WT_IN    = (_Float16*)(ws + (size_t)0);          // 4096x1024
    _Float16* WT_XPROJ = (_Float16*)(ws + (size_t)8388608);    // 96x2048
    _Float16* WT_DT    = (_Float16*)(ws + (size_t)8781824);    // 2048x64
    _Float16* WT_OUT   = (_Float16*)(ws + (size_t)9043968);    // 1024x2048
    _Float16* WT1      = (_Float16*)(ws + (size_t)13238272);   // 4096x1024
    _Float16* WT2      = (_Float16*)(ws + (size_t)21626880);   // 1024x4096
    _Float16* XH       = (_Float16*)(ws + (size_t)30015488);   // 8192x1024
    _Float16* XZH      = (_Float16*)(ws + (size_t)46792704);   // 8192x4096
    _Float16* XCH      = (_Float16*)(ws + (size_t)113901568);  // 8192x2048
    _Float16* PROJH    = (_Float16*)(ws + (size_t)147456000);  // 8192x96
    _Float16* DTH      = (_Float16*)(ws + (size_t)149028864);  // 8192x2048
    _Float16* YH       = (_Float16*)(ws + (size_t)182583296);  // 8192x2048
    float*    HRES     = (float*)   (ws + (size_t)216137728);  // 8192x1024 f32
    _Float16* HNH      = (_Float16*)(ws + (size_t)249692160);  // 8192x1024
    _Float16* GH       = (_Float16*)(ws + (size_t)266469376);  // 8192x4096

    const dim3 blk(256);
    #define CEIL256(x) ((unsigned)(((x) + 255) / 256))

    // ---- pre-pass: f16 weights (transposed to NxK) and f16 x ----
    cvt_transpose_f16_kernel<<<CEIL256(DDIM * 2 * EDIM), blk, 0, stream>>>(W_in,    WT_IN,    DDIM,  2 * EDIM);
    cvt_transpose_f16_kernel<<<CEIL256(EDIM * PROJW),    blk, 0, stream>>>(W_xproj, WT_XPROJ, EDIM,  PROJW);
    cvt_transpose_f16_kernel<<<CEIL256(DTRANK * EDIM),   blk, 0, stream>>>(W_dt,    WT_DT,    DTRANK, EDIM);
    cvt_transpose_f16_kernel<<<CEIL256(EDIM * DDIM),     blk, 0, stream>>>(W_out,   WT_OUT,   EDIM,  DDIM);
    cvt_transpose_f16_kernel<<<CEIL256(DDIM * HDIM),     blk, 0, stream>>>(W1,      WT1,      DDIM,  HDIM);
    cvt_transpose_f16_kernel<<<CEIL256(HDIM * DDIM),     blk, 0, stream>>>(W2,      WT2,      HDIM,  DDIM);
    cvt_f16_kernel<<<CEIL256(TOK * DDIM), blk, 0, stream>>>(x, XH, TOK * DDIM);

    // 1) xz = x @ W_in                       (8192 x 4096, K=1024)
    gemm_wmma_kernel<0, _Float16><<<dim3((2 * EDIM) / BN, TOK / BM), blk, 0, stream>>>(
        XH, WT_IN, nullptr, nullptr, XZH, TOK, 2 * EDIM, DDIM, DDIM, 2 * EDIM);

    // 2) xc = silu(conv(xp) + conv_b)
    conv_silu_kernel<<<CEIL256((size_t)BATCH * LSEQ * EDIM), blk, 0, stream>>>(
        XZH, conv_w, conv_b, XCH);

    // 3) proj = xc @ W_xproj                 (8192 x 96, K=2048; ragged N via TDM OOB)
    gemm_wmma_kernel<0, _Float16><<<dim3(1, TOK / BM), blk, 0, stream>>>(
        XCH, WT_XPROJ, nullptr, nullptr, PROJH, TOK, PROJW, EDIM, EDIM, PROJW);

    // 4) dt = softplus(proj[:, :64] @ W_dt + b_dt)   (8192 x 2048, K=64)
    gemm_wmma_kernel<3, _Float16><<<dim3(EDIM / BN, TOK / BM), blk, 0, stream>>>(
        PROJH, WT_DT, b_dt, nullptr, DTH, TOK, EDIM, DTRANK, PROJW, EDIM);

    // 5) selective scan + D*u + silu(z) gate -> Y
    ssm_scan_kernel<<<dim3(BATCH * (EDIM / 256)), blk, 0, stream>>>(
        DTH, PROJH, XCH, XZH, A_log, D_param, YH);

    // 6) h = x + Y @ W_out                   (8192 x 1024, K=2048) -> f32
    gemm_wmma_kernel<4, float><<<dim3(DDIM / BN, TOK / BM), blk, 0, stream>>>(
        YH, WT_OUT, nullptr, x, HRES, TOK, DDIM, EDIM, EDIM, DDIM);

    // 7) hn = LayerNorm(h)
    layernorm_kernel<<<dim3(TOK), blk, 0, stream>>>(HRES, ln_w, ln_b, HNH);

    // 8) g = gelu(hn @ W1 + b1)              (8192 x 4096, K=1024)
    gemm_wmma_kernel<2, _Float16><<<dim3(HDIM / BN, TOK / BM), blk, 0, stream>>>(
        HNH, WT1, b1, nullptr, GH, TOK, HDIM, DDIM, DDIM, HDIM);

    // 9) out = g @ W2 + b2                   (8192 x 1024, K=4096) -> f32 d_out
    gemm_wmma_kernel<1, float><<<dim3(DDIM / BN, TOK / BM), blk, 0, stream>>>(
        GH, WT2, b2, nullptr, out, TOK, DDIM, HDIM, HDIM, DDIM);
}
